// DiT_4758823763997
// MI455X (gfx1250) — matrix operand order repaired and checked by reading, not compiled, not run on hardware
//
#include <hip/hip_runtime.h>

// ---------------------------------------------------------------------------
// DiT forward for MI455X (gfx1250, wave32, WMMA 16x16x32 bf16).
// GEMMs: v_wmma_f32_16x16x32_bf16, LDS-staged B panel with async global->LDS
// copies (ASYNCcnt) and software-pipelined (double-buffered) K loop.
// ---------------------------------------------------------------------------

typedef __bf16 bf16_t;
typedef __attribute__((ext_vector_type(16))) __bf16 v16bf;
typedef __attribute__((ext_vector_type(8)))  __bf16 v8bf;
typedef __attribute__((ext_vector_type(8)))  float   v8f;
typedef int v4i_t __attribute__((vector_size(16)));   // matches builtin param type

#if defined(__HIP_DEVICE_COMPILE__) && __has_builtin(__builtin_amdgcn_global_load_async_to_lds_b128)
#define DIT_ASYNC 1
typedef __attribute__((address_space(1))) v4i_t* g_v4i_p;
typedef __attribute__((address_space(3))) v4i_t* l_v4i_p;
#else
#define DIT_ASYNC 0
#endif

// Model constants
#define BB_    8
#define DD_    768
#define NTOK_  256
#define MTOK_  2048        // B * N
#define NH_    12
#define HD_    64
#define MLP_   3072
#define ZBH_   96          // B * NH

// ---------------------------------------------------------------------------
// Fragment helpers (wave32; contiguous k order k = 16h + j, same for A and B)
// ---------------------------------------------------------------------------

__device__ inline v16bf zero_v16() {
  v16bf z;
#pragma unroll
  for (int i = 0; i < 16; ++i) z[i] = (__bf16)0.0f;
  return z;
}

// A fragment 16x32 bf16, row-major source [M,K]; lane l: m=l&15, h=l>>4;
// elem j -> k=k0+16h+j (16 contiguous bf16)  => two 16B loads.
__device__ inline v16bf load_a_frag(const bf16_t* __restrict__ A, int M, int K,
                                    int row0, int k0, int lane) {
  const int m = lane & 15, hh = lane >> 4;
  const int r = row0 + m;
  if (r >= M) return zero_v16();
  const bf16_t* p = A + (size_t)r * K + k0 + 16 * hh;
  union { v16bf v; v8bf h[2]; } u;
  u.h[0] = *(const v8bf*)(p);
  u.h[1] = *(const v8bf*)(p + 8);
  return u.v;
}

// B fragment from a transposed source Bt [Ncols, K] row-major:
// lane l: n=l&15, h=l>>4 ; elem j -> k = k0 + 16h + j (16 contiguous bf16).
__device__ inline v16bf load_bt_frag(const bf16_t* __restrict__ Bt, int K,
                                     int ncol0, int k0, int lane) {
  const int n = lane & 15, hh = lane >> 4;
  return *(const v16bf*)(Bt + (size_t)(ncol0 + n) * K + k0 + 16 * hh);
}

// ---------------------------------------------------------------------------
// Weight packing: fp32 [K,N] (or transposed [N,K]) -> bf16 B-fragment tiles
// Tile (nt,kt) = 32 lanes x 16 bf16, tiles ordered (nt, kt) with kt fastest.
// ---------------------------------------------------------------------------
__global__ void pack_w(const float* __restrict__ src, bf16_t* __restrict__ dst,
                       int K, int N, int transposed) {
  const int KT = K >> 5, NT = N >> 4;
  const int total = NT * KT * 32;
  int t = blockIdx.x * blockDim.x + threadIdx.x;
  if (t >= total) return;
  const int lane = t & 31;
  const int tile = t >> 5;
  const int kt = tile % KT, nt = tile / KT;
  const int n = nt * 16 + (lane & 15);
  const int kb = kt * 32 + 16 * (lane >> 4);
  bf16_t* d = dst + (size_t)t * 16;
#pragma unroll
  for (int j = 0; j < 16; ++j) {
    const int k = kb + j;
    const float v = transposed ? src[(size_t)n * K + k] : src[(size_t)k * N + n];
    d[j] = (bf16_t)v;
  }
}

// ---------------------------------------------------------------------------
// Stage one 8-tile B panel (8 KB) for K-step kt into LDS.
// 256 threads: thread -> (tile = tid>>5, 32B chunk = tid&31).
// ---------------------------------------------------------------------------
__device__ inline void stage_b_panel(const bf16_t* __restrict__ Bp, bf16_t* lds,
                                     int ntb, int kt, int KT, int tid) {
  const int tile = tid >> 5;
  const int chunk = tid & 31;
  const bf16_t* gp = Bp + (((size_t)(ntb + tile)) * KT + kt) * 512 + (size_t)chunk * 16;
  bf16_t* lp = lds + tile * 512 + chunk * 16;
#if DIT_ASYNC
  __builtin_amdgcn_global_load_async_to_lds_b128((g_v4i_p)gp, (l_v4i_p)lp, 0, 0);
  __builtin_amdgcn_global_load_async_to_lds_b128((g_v4i_p)(gp + 8), (l_v4i_p)(lp + 8), 0, 0);
#else
  v8bf t0 = *(const v8bf*)gp;
  v8bf t1 = *(const v8bf*)(gp + 8);
  *(v8bf*)lp = t0;
  *(v8bf*)(lp + 8) = t1;
#endif
  if (kt + 4 < KT)
    __builtin_prefetch((const void*)(gp + 2048), 0, 0);  // panel 4 K-steps ahead
}

__device__ inline void wait_async_lds() {
#if DIT_ASYNC
  asm volatile("s_wait_asynccnt 0x0" ::: "memory");
#endif
  __syncthreads();
}

// ---------------------------------------------------------------------------
// Packed-weight GEMM: C[M,N] = act(A_bf16[M,K] @ Bp + bias) (+ residual)
// 256 threads = 8 waves; wave tile 32x64 (2x4 WMMA frags); block tile 128x128.
// B panel double-buffered through LDS (async copies), A double-buffered in
// registers; K loop software-pipelined with unroll-2 ping-pong (KT even).
// N must be a multiple of 128; M guarded. ACT: 0=none 1=silu 2=gelu(exact)
// ---------------------------------------------------------------------------
template <int ACT, int RES, int OUTBF>
__global__ __launch_bounds__(256)
void gemm_wp(const bf16_t* __restrict__ A, const bf16_t* __restrict__ Bp,
             const float* __restrict__ bias, const float* __restrict__ resid,
             float* __restrict__ Cf, bf16_t* __restrict__ Cb,
             int M, int N, int K) {
  __shared__ bf16_t ldsb[2][8 * 512];
  const int tid  = threadIdx.x;
  const int lane = tid & 31;
  const int wave = tid >> 5;
  const int wr = wave >> 1, wc = wave & 1;
  const int row0 = blockIdx.y * 128 + wr * 32;
  const int ntb  = blockIdx.x * 8;          // block's first col tile
  const int nt0  = ntb + wc * 4;            // wave's first col tile
  const int KT = K >> 5;                    // always even here

  v8f acc[2][4];
  v8f vz = {};
#pragma unroll
  for (int mi = 0; mi < 2; ++mi)
#pragma unroll
    for (int ni = 0; ni < 4; ++ni) acc[mi][ni] = vz;

  v16bf a0[2], a1[2];
  stage_b_panel(Bp, &ldsb[0][0], ntb, 0, KT, tid);
  a0[0] = load_a_frag(A, M, K, row0,      0, lane);
  a0[1] = load_a_frag(A, M, K, row0 + 16, 0, lane);

  for (int kt = 0; kt < KT; kt += 2) {
    // ---- sub-iter kt: LDS buf 0, regs a0 ----
    wait_async_lds();
    v16bf bfr[4];
#pragma unroll
    for (int ni = 0; ni < 4; ++ni)
      bfr[ni] = *(const v16bf*)&ldsb[0][(size_t)((wc * 4 + ni) * 32 + lane) * 16];
    stage_b_panel(Bp, &ldsb[1][0], ntb, kt + 1, KT, tid);  // kt+1 < KT (KT even)
    a1[0] = load_a_frag(A, M, K, row0,      (kt + 1) * 32, lane);
    a1[1] = load_a_frag(A, M, K, row0 + 16, (kt + 1) * 32, lane);
#pragma unroll
    for (int ni = 0; ni < 4; ++ni) {
      acc[0][ni] = __builtin_amdgcn_wmma_f32_16x16x32_bf16(
          false, a0[0], false, bfr[ni], (short)0, acc[0][ni], false, false);
      acc[1][ni] = __builtin_amdgcn_wmma_f32_16x16x32_bf16(
          false, a0[1], false, bfr[ni], (short)0, acc[1][ni], false, false);
    }
    // ---- sub-iter kt+1: LDS buf 1, regs a1 ----
    wait_async_lds();
    v16bf bfr2[4];
#pragma unroll
    for (int ni = 0; ni < 4; ++ni)
      bfr2[ni] = *(const v16bf*)&ldsb[1][(size_t)((wc * 4 + ni) * 32 + lane) * 16];
    if (kt + 2 < KT) {
      stage_b_panel(Bp, &ldsb[0][0], ntb, kt + 2, KT, tid);
      a0[0] = load_a_frag(A, M, K, row0,      (kt + 2) * 32, lane);
      a0[1] = load_a_frag(A, M, K, row0 + 16, (kt + 2) * 32, lane);
    }
#pragma unroll
    for (int ni = 0; ni < 4; ++ni) {
      acc[0][ni] = __builtin_amdgcn_wmma_f32_16x16x32_bf16(
          false, a1[0], false, bfr2[ni], (short)0, acc[0][ni], false, false);
      acc[1][ni] = __builtin_amdgcn_wmma_f32_16x16x32_bf16(
          false, a1[1], false, bfr2[ni], (short)0, acc[1][ni], false, false);
    }
  }

  const int hh = lane >> 4, ln = lane & 15;
#pragma unroll
  for (int mi = 0; mi < 2; ++mi) {
#pragma unroll
    for (int ni = 0; ni < 4; ++ni) {
      const int col = (nt0 + ni) * 16 + ln;
#pragma unroll
      for (int r = 0; r < 8; ++r) {
        const int row = row0 + mi * 16 + r + 8 * hh;
        if (row < M) {
          float v = acc[mi][ni][r] + bias[col];
          if (ACT == 1) v = v / (1.0f + __expf(-v));
          else if (ACT == 2) v = 0.5f * v * (1.0f + erff(v * 0.70710678118654752f));
          if (RES) v += resid[(size_t)row * N + col];
          if (OUTBF) Cb[(size_t)row * N + col] = (bf16_t)v;
          else       Cf[(size_t)row * N + col] = v;
        }
      }
    }
  }
}

// ---------------------------------------------------------------------------
// Batched GEMM with transposed-B source (attention): per z,
//   C = scale * (A[z] @ Bt[z]^T),  A:[M,K] bf16, Bt:[N,K] bf16.
// 8 waves cover 256 rows; wave tile 32x64; register double-buffered K loop.
// M%256==0, N%64==0, KT even.
// MODE 0: C fp32 contiguous [z][M][N] (scores)
// MODE 1: C bf16 scattered into [B,NTOK,D] at head offset (attention output)
// ---------------------------------------------------------------------------
template <int MODE, int OUTBF>
__global__ __launch_bounds__(256)
void gemm_bt(const bf16_t* __restrict__ A, const bf16_t* __restrict__ Bt,
             float* __restrict__ Cf, bf16_t* __restrict__ Cb,
             int M, int N, int K, float scale) {
  const int z = blockIdx.z;
  const bf16_t* Az = A + (size_t)z * M * K;
  const bf16_t* Bz = Bt + (size_t)z * N * K;
  const int lane = threadIdx.x & 31;
  const int wave = threadIdx.x >> 5;
  const int row0 = blockIdx.y * 256 + wave * 32;
  const int col0 = blockIdx.x * 64;
  const int KT = K >> 5;                 // 2 or 8 -> even

  v8f acc[2][4];
  v8f vz = {};
#pragma unroll
  for (int mi = 0; mi < 2; ++mi)
#pragma unroll
    for (int ni = 0; ni < 4; ++ni) acc[mi][ni] = vz;

  v16bf a0[2], b0[4], a1[2], b1[4];
  a0[0] = load_a_frag(Az, M, K, row0,      0, lane);
  a0[1] = load_a_frag(Az, M, K, row0 + 16, 0, lane);
#pragma unroll
  for (int ni = 0; ni < 4; ++ni) b0[ni] = load_bt_frag(Bz, K, col0 + ni * 16, 0, lane);

  for (int kt = 0; kt < KT; kt += 2) {
    a1[0] = load_a_frag(Az, M, K, row0,      (kt + 1) * 32, lane);
    a1[1] = load_a_frag(Az, M, K, row0 + 16, (kt + 1) * 32, lane);
#pragma unroll
    for (int ni = 0; ni < 4; ++ni)
      b1[ni] = load_bt_frag(Bz, K, col0 + ni * 16, (kt + 1) * 32, lane);
#pragma unroll
    for (int ni = 0; ni < 4; ++ni) {
      acc[0][ni] = __builtin_amdgcn_wmma_f32_16x16x32_bf16(
          false, a0[0], false, b0[ni], (short)0, acc[0][ni], false, false);
      acc[1][ni] = __builtin_amdgcn_wmma_f32_16x16x32_bf16(
          false, a0[1], false, b0[ni], (short)0, acc[1][ni], false, false);
    }
    if (kt + 2 < KT) {
      a0[0] = load_a_frag(Az, M, K, row0,      (kt + 2) * 32, lane);
      a0[1] = load_a_frag(Az, M, K, row0 + 16, (kt + 2) * 32, lane);
#pragma unroll
      for (int ni = 0; ni < 4; ++ni)
        b0[ni] = load_bt_frag(Bz, K, col0 + ni * 16, (kt + 2) * 32, lane);
    }
#pragma unroll
    for (int ni = 0; ni < 4; ++ni) {
      acc[0][ni] = __builtin_amdgcn_wmma_f32_16x16x32_bf16(
          false, a1[0], false, b1[ni], (short)0, acc[0][ni], false, false);
      acc[1][ni] = __builtin_amdgcn_wmma_f32_16x16x32_bf16(
          false, a1[1], false, b1[ni], (short)0, acc[1][ni], false, false);
    }
  }

  const int hh = lane >> 4, ln = lane & 15;
#pragma unroll
  for (int mi = 0; mi < 2; ++mi) {
#pragma unroll
    for (int ni = 0; ni < 4; ++ni) {
      const int col = col0 + ni * 16 + ln;
#pragma unroll
      for (int r = 0; r < 8; ++r) {
        const int row = row0 + mi * 16 + r + 8 * hh;
        const float v = acc[mi][ni][r] * scale;
        if (MODE == 0) {
          Cf[(size_t)z * M * N + (size_t)row * N + col] = v;
        } else {
          const int b = z / NH_, head = z % NH_;
          Cb[((size_t)(b * NTOK_ + row)) * DD_ + head * HD_ + col] = (bf16_t)v;
        }
      }
    }
  }
}

// ---------------------------------------------------------------------------
// Block reduction helper (256 threads = 8 waves)
// ---------------------------------------------------------------------------
__device__ inline float blk_sum(float v, volatile float* s) {
#pragma unroll
  for (int off = 16; off > 0; off >>= 1) v += __shfl_xor(v, off, 32);
  __syncthreads();
  if ((threadIdx.x & 31) == 0) s[threadIdx.x >> 5] = v;
  __syncthreads();
  float tot = 0.f;
  const int nw = blockDim.x >> 5;
  for (int i = 0; i < nw; ++i) tot += s[i];
  return tot;
}

// ---------------------------------------------------------------------------
// Elementwise / normalization kernels
// ---------------------------------------------------------------------------

__global__ void temb0_kernel(const float* __restrict__ T, bf16_t* __restrict__ O) {
  int i = blockIdx.x * blockDim.x + threadIdx.x;
  if (i >= BB_ * DD_) return;
  const int b = i / DD_, d = i % DD_;
  const float ts = T[b] * 1000.0f;
  const int j = (d < 384) ? d : d - 384;
  const float freq = __expf(-9.2103403719761836f * (float)j / 383.0f);
  const float e = ts * freq;
  O[i] = (bf16_t)((d < 384) ? __sinf(e) : __cosf(e));
}

__global__ void silu_to_bf16(const float* __restrict__ X, bf16_t* __restrict__ Y, int n) {
  int i = blockIdx.x * blockDim.x + threadIdx.x;
  if (i >= n) return;
  const float v = X[i];
  Y[i] = (bf16_t)(v / (1.0f + __expf(-v)));
}

// im2col: x[B,3,256,256] -> patches-as-rows bf16 [2048, 768] (k = c*256+p*16+q)
__global__ void im2col_kernel(const float* __restrict__ X, bf16_t* __restrict__ XP) {
  int i = blockIdx.x * blockDim.x + threadIdx.x;
  if (i >= MTOK_ * DD_) return;
  const int kcol = i % DD_;
  const int tok  = i / DD_;
  const int b = tok >> 8, n = tok & 255;
  const int hp = n >> 4, wp = n & 15;
  const int c = kcol / 256;
  const int rem = kcol % 256;
  const int p = rem >> 4, q = rem & 15;
  XP[i] = (bf16_t)X[(((size_t)b * 3 + c) * 256 + (hp * 16 + p)) * 256 + (wp * 16 + q)];
}

// GroupNorm(8) over patches [b,n,d] per (b,group): mean/var over 96 ch x 256 tok
__global__ void groupnorm_kernel(const float* __restrict__ P,
                                 const float* __restrict__ g, const float* __restrict__ b,
                                 float* __restrict__ H) {
  __shared__ float s[8];
  const int blk = blockIdx.x;       // b*8 + grp
  const int bb = blk >> 3, grp = blk & 7;
  const int tid = threadIdx.x;
  float sum = 0.f, sq = 0.f;
  for (int idx = tid; idx < 96 * 256; idx += 256) {
    const int ch = idx >> 8, n = idx & 255;
    const float v = P[((size_t)(bb * 256 + n)) * DD_ + grp * 96 + ch];
    sum += v; sq += v * v;
  }
  const float tot  = blk_sum(sum, s);
  const float tot2 = blk_sum(sq, s);
  const float mean = tot * (1.0f / 24576.0f);
  const float var  = tot2 * (1.0f / 24576.0f) - mean * mean;
  const float rstd = rsqrtf(var + 1e-5f);
  for (int idx = tid; idx < 96 * 256; idx += 256) {
    const int ch = idx >> 8, n = idx & 255;
    const int d = grp * 96 + ch;
    const size_t o = ((size_t)(bb * 256 + n)) * DD_ + d;
    H[o] = (P[o] - mean) * rstd * g[d] + b[d];
  }
}

// adaLN modulation + LN1: hmod = ln(h)*(1+scale)+shift ; hn = ln(hmod)*g1+b1 (bf16)
__global__ void adaln_mod_kernel(const float* __restrict__ H, const float* __restrict__ SC,
                                 const float* __restrict__ g1, const float* __restrict__ b1,
                                 float* __restrict__ Hmod, bf16_t* __restrict__ Hn) {
  __shared__ float s[8];
  const int tok = blockIdx.x;
  const int bb = tok >> 8;
  const int tid = threadIdx.x;
  const float* row = H + (size_t)tok * DD_;
  float x0 = row[tid], x1 = row[tid + 256], x2 = row[tid + 512];
  const float mean = blk_sum(x0 + x1 + x2, s) * (1.0f / 768.0f);
  const float d0 = x0 - mean, d1 = x1 - mean, d2 = x2 - mean;
  const float var = blk_sum(d0 * d0 + d1 * d1 + d2 * d2, s) * (1.0f / 768.0f);
  const float rstd = rsqrtf(var + 1e-6f);
  const float* shift = SC + (size_t)bb * 1536;
  const float* scl   = shift + 768;
  float hm[3];
  const float dv[3] = {d0, d1, d2};
#pragma unroll
  for (int i = 0; i < 3; ++i) {
    const int d = tid + i * 256;
    hm[i] = dv[i] * rstd * (1.0f + scl[d]) + shift[d];
    Hmod[(size_t)tok * DD_ + d] = hm[i];
  }
  const float mean2 = blk_sum(hm[0] + hm[1] + hm[2], s) * (1.0f / 768.0f);
  const float e0 = hm[0] - mean2, e1 = hm[1] - mean2, e2 = hm[2] - mean2;
  const float var2 = blk_sum(e0 * e0 + e1 * e1 + e2 * e2, s) * (1.0f / 768.0f);
  const float rstd2 = rsqrtf(var2 + 1e-6f);
  const float ev[3] = {e0, e1, e2};
#pragma unroll
  for (int i = 0; i < 3; ++i) {
    const int d = tid + i * 256;
    Hn[(size_t)tok * DD_ + d] = (bf16_t)(ev[i] * rstd2 * g1[d] + b1[d]);
  }
}

// generic LN+affine -> bf16
__global__ void ln_affine_kernel(const float* __restrict__ X, const float* __restrict__ g,
                                 const float* __restrict__ b, bf16_t* __restrict__ Y) {
  __shared__ float s[8];
  const int tok = blockIdx.x;
  const int tid = threadIdx.x;
  const float* row = X + (size_t)tok * DD_;
  float x0 = row[tid], x1 = row[tid + 256], x2 = row[tid + 512];
  const float mean = blk_sum(x0 + x1 + x2, s) * (1.0f / 768.0f);
  const float d0 = x0 - mean, d1 = x1 - mean, d2 = x2 - mean;
  const float var = blk_sum(d0 * d0 + d1 * d1 + d2 * d2, s) * (1.0f / 768.0f);
  const float rstd = rsqrtf(var + 1e-6f);
  bf16_t* yr = Y + (size_t)tok * DD_;
  yr[tid]       = (bf16_t)(d0 * rstd * g[tid]       + b[tid]);
  yr[tid + 256] = (bf16_t)(d1 * rstd * g[tid + 256] + b[tid + 256]);
  yr[tid + 512] = (bf16_t)(d2 * rstd * g[tid + 512] + b[tid + 512]);
}

// rotary on q,k; stage Qr/Kr bf16 [z][256][64], Vt bf16 [z][64][256]
__global__ void rope_prep_kernel(const float* __restrict__ Q, const float* __restrict__ Kk,
                                 const float* __restrict__ V,
                                 bf16_t* __restrict__ Qr, bf16_t* __restrict__ Kr,
                                 bf16_t* __restrict__ Vt) {
  int i = blockIdx.x * blockDim.x + threadIdx.x;
  if (i >= ZBH_ * NTOK_ * HD_) return;
  const int d = i & 63;
  const int n = (i >> 6) & 255;
  const int z = i >> 14;
  const int bb = z / NH_, head = z % NH_;
  const size_t src = ((size_t)(bb * NTOK_ + n)) * DD_ + head * HD_ + d;
  const int j = (d < 32) ? d : d - 32;
  const float inv = __expf(-9.2103403719761836f * (float)j / 32.0f);
  const float ang = (float)n * inv;
  const float c = __cosf(ang), sn = __sinf(ang);
  const size_t src2 = (d < 32) ? src + 32 : src - 32;
  const float qs = (d < 32) ? -Q[src2]  : Q[src2];
  const float ks = (d < 32) ? -Kk[src2] : Kk[src2];
  const size_t zo = (size_t)z * (NTOK_ * HD_);
  Qr[zo + (size_t)n * HD_ + d] = (bf16_t)(Q[src] * c + qs * sn);
  Kr[zo + (size_t)n * HD_ + d] = (bf16_t)(Kk[src] * c + ks * sn);
  Vt[zo + (size_t)d * NTOK_ + n] = (bf16_t)V[src];
}

// softmax over rows of S [rows][256] -> bf16 P ; one wave per row
__global__ void softmax_rows_kernel(const float* __restrict__ S, bf16_t* __restrict__ P) {
  const int row = blockIdx.x * 8 + (threadIdx.x >> 5);
  const int lane = threadIdx.x & 31;
  const float* r = S + (size_t)row * 256;
  float v[8];
  float mx = -3.0e38f;
#pragma unroll
  for (int i = 0; i < 8; ++i) { v[i] = r[i * 32 + lane]; mx = fmaxf(mx, v[i]); }
#pragma unroll
  for (int off = 16; off > 0; off >>= 1) mx = fmaxf(mx, __shfl_xor(mx, off, 32));
  float sum = 0.f;
#pragma unroll
  for (int i = 0; i < 8; ++i) { v[i] = __expf(v[i] - mx); sum += v[i]; }
#pragma unroll
  for (int off = 16; off > 0; off >>= 1) sum += __shfl_xor(sum, off, 32);
  const float inv = 1.0f / sum;
  bf16_t* p = P + (size_t)row * 256;
#pragma unroll
  for (int i = 0; i < 8; ++i) p[i * 32 + lane] = (bf16_t)(v[i] * inv);
}

// final: F [2048,2304] -> out [B,3,256,256]
__global__ void unpatchify_kernel(const float* __restrict__ F, float* __restrict__ OUT) {
  int i = blockIdx.x * blockDim.x + threadIdx.x;
  if (i >= BB_ * 3 * 256 * 256) return;
  const int w = i & 255;
  const int h = (i >> 8) & 255;
  const int rest = i >> 16;
  const int c = rest % 3;
  const int bb = rest / 3;
  const int hp = h >> 4, p = h & 15, wp = w >> 4, q = w & 15;
  const int tok = bb * NTOK_ + hp * 16 + wp;
  const int col = c * 256 + p * 16 + q;
  OUT[i] = F[(size_t)tok * 2304 + col];
}

// ---------------------------------------------------------------------------
// Host orchestration
// ---------------------------------------------------------------------------
extern "C" void kernel_launch(void* const* d_in, const int* in_sizes, int n_in,
                              void* d_out, int out_size, void* d_ws, size_t ws_size,
                              hipStream_t stream) {
  (void)in_sizes; (void)n_in; (void)out_size; (void)ws_size;
  const float* x      = (const float*)d_in[0];
  const float* t      = (const float*)d_in[1];
  const float* conv_w = (const float*)d_in[2];
  const float* conv_b = (const float*)d_in[3];
  const float* gn_g   = (const float*)d_in[4];
  const float* gn_b   = (const float*)d_in[5];
  const float* t_w1   = (const float*)d_in[6];
  const float* t_b1   = (const float*)d_in[7];
  const float* t_w2   = (const float*)d_in[8];
  const float* t_b2   = (const float*)d_in[9];
  const float* ada_w  = (const float*)d_in[10];
  const float* ada_b  = (const float*)d_in[11];
  const float* ln1_g  = (const float*)d_in[12];
  const float* ln1_b  = (const float*)d_in[13];
  const float* ln2_g  = (const float*)d_in[14];
  const float* ln2_b  = (const float*)d_in[15];
  const float* wq     = (const float*)d_in[16];
  const float* bq     = (const float*)d_in[17];
  const float* wk     = (const float*)d_in[18];
  const float* bk     = (const float*)d_in[19];
  const float* wv     = (const float*)d_in[20];
  const float* bv     = (const float*)d_in[21];
  const float* wo     = (const float*)d_in[22];
  const float* bo     = (const float*)d_in[23];
  const float* w1     = (const float*)d_in[24];
  const float* b1     = (const float*)d_in[25];
  const float* w2     = (const float*)d_in[26];
  const float* b2     = (const float*)d_in[27];
  const float* fin_g  = (const float*)d_in[28];
  const float* fin_b  = (const float*)d_in[29];
  const float* out_w  = (const float*)d_in[30];
  const float* out_b  = (const float*)d_in[31];
  float* out = (float*)d_out;

  char* base = (char*)d_ws;
  size_t off = 0;
  auto alloc = [&](size_t bytes) -> void* {
    void* p = base + off;
    off = (off + bytes + 255) & ~(size_t)255;
    return p;
  };

  // packed weights (bf16)
  bf16_t* pk_tw1  = (bf16_t*)alloc((size_t)768 * 768 * 2);
  bf16_t* pk_tw2  = (bf16_t*)alloc((size_t)768 * 768 * 2);
  bf16_t* pk_conv = (bf16_t*)alloc((size_t)768 * 768 * 2);
  bf16_t* pk_outw = (bf16_t*)alloc((size_t)768 * 2304 * 2);
  bf16_t* pk_ada  = (bf16_t*)alloc((size_t)768 * 1536 * 2);
  bf16_t* pk_wq   = (bf16_t*)alloc((size_t)768 * 768 * 2);
  bf16_t* pk_wk   = (bf16_t*)alloc((size_t)768 * 768 * 2);
  bf16_t* pk_wv   = (bf16_t*)alloc((size_t)768 * 768 * 2);
  bf16_t* pk_wo   = (bf16_t*)alloc((size_t)768 * 768 * 2);
  bf16_t* pk_w1   = (bf16_t*)alloc((size_t)768 * 3072 * 2);
  bf16_t* pk_w2   = (bf16_t*)alloc((size_t)3072 * 768 * 2);
  // activations
  bf16_t* temb0_bf = (bf16_t*)alloc((size_t)BB_ * DD_ * 2);
  bf16_t* t1_bf    = (bf16_t*)alloc((size_t)BB_ * DD_ * 2);
  float*  temb_f   = (float*)alloc((size_t)BB_ * DD_ * 4);
  bf16_t* stemb_bf = (bf16_t*)alloc((size_t)BB_ * DD_ * 2);
  float*  sc_f     = (float*)alloc((size_t)BB_ * 1536 * 4);
  bf16_t* xp_bf    = (bf16_t*)alloc((size_t)MTOK_ * DD_ * 2);
  float*  patch_f  = (float*)alloc((size_t)MTOK_ * DD_ * 4);
  float*  h_f      = (float*)alloc((size_t)MTOK_ * DD_ * 4);
  float*  hmod_f   = (float*)alloc((size_t)MTOK_ * DD_ * 4);
  float*  h1_f     = (float*)alloc((size_t)MTOK_ * DD_ * 4);
  bf16_t* hn_bf    = (bf16_t*)alloc((size_t)MTOK_ * DD_ * 2);
  bf16_t* hn2_bf   = (bf16_t*)alloc((size_t)MTOK_ * DD_ * 2);
  float*  q_f      = (float*)alloc((size_t)MTOK_ * DD_ * 4);
  float*  k_f      = (float*)alloc((size_t)MTOK_ * DD_ * 4);
  float*  v_f      = (float*)alloc((size_t)MTOK_ * DD_ * 4);
  bf16_t* Qr       = (bf16_t*)alloc((size_t)ZBH_ * NTOK_ * HD_ * 2);
  bf16_t* Kr       = (bf16_t*)alloc((size_t)ZBH_ * NTOK_ * HD_ * 2);
  bf16_t* Vt       = (bf16_t*)alloc((size_t)ZBH_ * NTOK_ * HD_ * 2);
  float*  S_f      = (float*)alloc((size_t)ZBH_ * NTOK_ * NTOK_ * 4);
  bf16_t* P_bf     = (bf16_t*)alloc((size_t)ZBH_ * NTOK_ * NTOK_ * 2);
  bf16_t* attn_bf  = (bf16_t*)alloc((size_t)MTOK_ * DD_ * 2);
  bf16_t* mlp1_bf  = (bf16_t*)alloc((size_t)MTOK_ * MLP_ * 2);
  float*  final_f  = (float*)alloc((size_t)MTOK_ * 2304 * 4);
  bf16_t* fin_bf   = (bf16_t*)alloc((size_t)MTOK_ * DD_ * 2);

  const dim3 blk(256);
  auto pack_grid = [](int K, int N) {
    const int total = (N / 16) * (K / 32) * 32;
    return dim3((unsigned)((total + 255) / 256));
  };
  auto gemm_grid = [](int M, int N) {
    return dim3((unsigned)(N / 128), (unsigned)((M + 127) / 128));
  };

  // --- one-time packs ---
  pack_w<<<pack_grid(768, 768), blk, 0, stream>>>(t_w1, pk_tw1, 768, 768, 0);
  pack_w<<<pack_grid(768, 768), blk, 0, stream>>>(t_w2, pk_tw2, 768, 768, 0);
  pack_w<<<pack_grid(768, 768), blk, 0, stream>>>(conv_w, pk_conv, 768, 768, 1);
  pack_w<<<pack_grid(768, 2304), blk, 0, stream>>>(out_w, pk_outw, 768, 2304, 0);

  // --- time embedding ---
  temb0_kernel<<<dim3((BB_ * DD_ + 255) / 256), blk, 0, stream>>>(t, temb0_bf);
  gemm_wp<1, 0, 1><<<gemm_grid(BB_, 768), blk, 0, stream>>>(
      temb0_bf, pk_tw1, t_b1, nullptr, nullptr, t1_bf, BB_, 768, 768);
  gemm_wp<0, 0, 0><<<gemm_grid(BB_, 768), blk, 0, stream>>>(
      t1_bf, pk_tw2, t_b2, nullptr, temb_f, nullptr, BB_, 768, 768);
  silu_to_bf16<<<dim3((BB_ * DD_ + 255) / 256), blk, 0, stream>>>(temb_f, stemb_bf, BB_ * DD_);

  // --- patch embed + GroupNorm ---
  im2col_kernel<<<dim3((MTOK_ * DD_ + 255) / 256), blk, 0, stream>>>(x, xp_bf);
  gemm_wp<0, 0, 0><<<gemm_grid(MTOK_, 768), blk, 0, stream>>>(
      xp_bf, pk_conv, conv_b, nullptr, patch_f, nullptr, MTOK_, 768, 768);
  groupnorm_kernel<<<dim3(BB_ * 8), blk, 0, stream>>>(patch_f, gn_g, gn_b, h_f);

  // --- transformer layers ---
  for (int i = 0; i < 12; ++i) {
    const size_t wofs  = (size_t)i * 768 * 768;
    const size_t mofs1 = (size_t)i * 768 * 3072;
    pack_w<<<pack_grid(768, 1536), blk, 0, stream>>>(ada_w + (size_t)i * 768 * 1536, pk_ada, 768, 1536, 0);
    pack_w<<<pack_grid(768, 768), blk, 0, stream>>>(wq + wofs, pk_wq, 768, 768, 0);
    pack_w<<<pack_grid(768, 768), blk, 0, stream>>>(wk + wofs, pk_wk, 768, 768, 0);
    pack_w<<<pack_grid(768, 768), blk, 0, stream>>>(wv + wofs, pk_wv, 768, 768, 0);
    pack_w<<<pack_grid(768, 768), blk, 0, stream>>>(wo + wofs, pk_wo, 768, 768, 0);
    pack_w<<<pack_grid(768, 3072), blk, 0, stream>>>(w1 + mofs1, pk_w1, 768, 3072, 0);
    pack_w<<<pack_grid(3072, 768), blk, 0, stream>>>(w2 + mofs1, pk_w2, 3072, 768, 0);

    // adaLN-zero modulation
    gemm_wp<0, 0, 0><<<gemm_grid(BB_, 1536), blk, 0, stream>>>(
        stemb_bf, pk_ada, ada_b + (size_t)i * 1536, nullptr, sc_f, nullptr, BB_, 1536, 768);
    adaln_mod_kernel<<<dim3(MTOK_), blk, 0, stream>>>(
        h_f, sc_f, ln1_g + (size_t)i * 768, ln1_b + (size_t)i * 768, hmod_f, hn_bf);

    // QKV
    gemm_wp<0, 0, 0><<<gemm_grid(MTOK_, 768), blk, 0, stream>>>(
        hn_bf, pk_wq, bq + (size_t)i * 768, nullptr, q_f, nullptr, MTOK_, 768, 768);
    gemm_wp<0, 0, 0><<<gemm_grid(MTOK_, 768), blk, 0, stream>>>(
        hn_bf, pk_wk, bk + (size_t)i * 768, nullptr, k_f, nullptr, MTOK_, 768, 768);
    gemm_wp<0, 0, 0><<<gemm_grid(MTOK_, 768), blk, 0, stream>>>(
        hn_bf, pk_wv, bv + (size_t)i * 768, nullptr, v_f, nullptr, MTOK_, 768, 768);

    // rotary + staging
    rope_prep_kernel<<<dim3((ZBH_ * NTOK_ * HD_ + 255) / 256), blk, 0, stream>>>(
        q_f, k_f, v_f, Qr, Kr, Vt);

    // attention: S = Q K^T * scale ; P = softmax(S) ; O = P V
    gemm_bt<0, 0><<<dim3(4, 1, ZBH_), blk, 0, stream>>>(
        Qr, Kr, S_f, nullptr, NTOK_, NTOK_, HD_, 0.125f);
    softmax_rows_kernel<<<dim3(ZBH_ * NTOK_ / 8), blk, 0, stream>>>(S_f, P_bf);
    gemm_bt<1, 1><<<dim3(1, 1, ZBH_), blk, 0, stream>>>(
        P_bf, Vt, nullptr, attn_bf, NTOK_, HD_, NTOK_, 1.0f);

    // projection + residual
    gemm_wp<0, 1, 0><<<gemm_grid(MTOK_, 768), blk, 0, stream>>>(
        attn_bf, pk_wo, bo + (size_t)i * 768, hmod_f, h1_f, nullptr, MTOK_, 768, 768);

    // MLP
    ln_affine_kernel<<<dim3(MTOK_), blk, 0, stream>>>(
        h1_f, ln2_g + (size_t)i * 768, ln2_b + (size_t)i * 768, hn2_bf);
    gemm_wp<2, 0, 1><<<gemm_grid(MTOK_, 3072), blk, 0, stream>>>(
        hn2_bf, pk_w1, b1 + (size_t)i * 3072, nullptr, nullptr, mlp1_bf, MTOK_, 3072, 768);
    gemm_wp<0, 1, 0><<<gemm_grid(MTOK_, 768), blk, 0, stream>>>(
        mlp1_bf, pk_w2, b2 + (size_t)i * 768, h1_f, h_f, nullptr, MTOK_, 768, 3072);
  }

  // --- final head ---
  ln_affine_kernel<<<dim3(MTOK_), blk, 0, stream>>>(h_f, fin_g, fin_b, fin_bf);
  gemm_wp<0, 0, 0><<<gemm_grid(MTOK_, 2304), blk, 0, stream>>>(
      fin_bf, pk_outw, out_b, nullptr, final_f, nullptr, MTOK_, 2304, 768);
  unpatchify_kernel<<<dim3((BB_ * 3 * 256 * 256 + 255) / 256), blk, 0, stream>>>(final_f, out);
}
